// trans_encoder_44092134261544
// MI455X (gfx1250) — matrix operand order repaired and checked by reading, not compiled, not run on hardware
//
#include <hip/hip_runtime.h>
#include <cstdint>

// ---------------------------------------------------------------------------
// CDNA5 (gfx1250) wave32 WMMA implementation of the linear-attention encoder.
// GEMMs: v_wmma_f32_16x16x32_bf16 (bf16 in, f32 accumulate).
// A-tile staging: global_load_async_to_lds_b128 (ASYNCcnt) + global_prefetch.
// B-tile staging: register-pipelined global loads + transposed LDS scatter.
// ---------------------------------------------------------------------------

typedef __attribute__((ext_vector_type(16))) __bf16 v16bf;
typedef __attribute__((ext_vector_type(8)))  __bf16 v8bf;
typedef __attribute__((ext_vector_type(8)))  float  v8f;

union FragBF { v16bf v; v8bf h[2]; };

#define TILE_M 128
#define TILE_N 128
#define TILE_K 64
#define LDSS   72   // padded LDS row stride (bf16 elems); 72*2=144B = 9*16B

// Problem constants (B=2, L=8192, D=1024, H=8, d=128)
#define C_B 2
#define C_L 8192
#define C_D 1024
#define C_H 8

// low 32 bits of a generic pointer to __shared__ == logical LDS byte address
__device__ __forceinline__ unsigned lds_addr32(const void* p)
{
    return (unsigned)(uintptr_t)p;
}

// ---------------------------------------------------------------------------
// Generic bf16 WMMA GEMM:  C = A(MxK) * B(KxN), row-major, dims % 128 == 0,
// K % 64 == 0.  Optional batching (blockIdx.z): z -> (b = z/HH, h = z%HH),
// A/C offset = b*strideAb + h*strideAh, B offset = z*strideBz.
// Epilogues:
//   0: outF[idx] = c * sc            (f32)
//   1: outB[idx] = bf16(max(c,0))    (relu, bf16)
//   2: outB[idx] = bf16(elu(c)+1); outB2[idx] = bf16(c*invL)   (feat / V)
//   3: outB[idx] = bf16(c * sc)      (bf16, per-row scale for msg)
// sc = rowscale ? rowscale[row]*scalemul : scalemul
// ---------------------------------------------------------------------------
__global__ __launch_bounds__(256) void gemm_bf16_kernel(
    const __bf16* __restrict__ A, int lda, long long strideAb, long long strideAh,
    const __bf16* __restrict__ Bm, int ldb, long long strideBz,
    float* __restrict__ outF, __bf16* __restrict__ outB, __bf16* __restrict__ outB2,
    int ldc, long long strideCb, long long strideCh,
    const float* __restrict__ rowscale, long long strideRSz,
    float scalemul, float invL, int HH, int K, int epi)
{
    __shared__ __bf16 As[TILE_M * LDSS];   // A tile, row-major [m][k]
    __shared__ __bf16 Bs[TILE_N * LDSS];   // B tile TRANSPOSED [n][k]

    const int tid  = threadIdx.x;
    const int lane = tid & 31;
    const int wid  = tid >> 5;
    const int wm   = wid & 3;          // 4 waves over M (32 rows each)
    const int wn   = wid >> 2;         // 2 waves over N (64 cols each)
    const int lr   = lane & 15;
    const int g    = lane >> 4;        // lane group (0 or 1)

    const int rowBase = blockIdx.x * TILE_M;
    const int n0      = blockIdx.y * TILE_N;

    const int z  = blockIdx.z;
    const int bz = z / HH;
    const int hz = z - bz * HH;
    A  += (long long)bz * strideAb + (long long)hz * strideAh;
    Bm += (long long)z * strideBz;
    const long long cOff = (long long)bz * strideCb + (long long)hz * strideCh;
    const float* rs = rowscale ? rowscale + (long long)z * strideRSz : nullptr;

    // per-thread staging coordinates
    const int ra = (tid * 4) >> 5;          // not used; kept simple below
    (void)ra;

    v8f acc[2][4] = {};

    const int T = K / TILE_K;

    // --- prefetch B tile 0 into registers ---
    uint4 breg[4];
    #pragma unroll
    for (int i = 0; i < 4; ++i) {
        int v = tid + i * 256;
        int r = v >> 4;                   // k row 0..63
        int c = (v & 15) * 8;             // n col 0..120
        breg[i] = *(const uint4*)(Bm + (long long)r * ldb + (n0 + c));
    }

    for (int t = 0; t < T; ++t) {
        const int k0 = t * TILE_K;
        __syncthreads();                  // LDS free from previous compute

        // --- A tile (128 x 64): async copy global -> LDS (ASYNCcnt) ---
        #pragma unroll
        for (int i = 0; i < 4; ++i) {
            int v = tid + i * 256;        // 1024 slots of 8 bf16 (16B)
            int r = v >> 3;               // row 0..127
            int c = (v & 7) * 8;          // col 0..56
            unsigned long long ga =
                (unsigned long long)(A + (long long)(rowBase + r) * lda + (k0 + c));
            unsigned la = lds_addr32(&As[r * LDSS + c]);
            asm volatile("global_load_async_to_lds_b128 %0, %1, off"
                         :: "v"(la), "v"(ga)
                         : "memory");
        }

        // --- scatter prefetched B tile transposed into LDS [n][k] ---
        #pragma unroll
        for (int i = 0; i < 4; ++i) {
            int v = tid + i * 256;
            int r = v >> 4;
            int c = (v & 15) * 8;
            const __bf16* e = (const __bf16*)&breg[i];
            #pragma unroll
            for (int j = 0; j < 8; ++j)
                Bs[(c + j) * LDSS + r] = e[j];
        }

        // --- prefetch B tile t+1 into registers; L2-prefetch A tile t+1 ---
        if (t + 1 < T) {
            const int kn = k0 + TILE_K;
            #pragma unroll
            for (int i = 0; i < 4; ++i) {
                int v = tid + i * 256;
                int r = v >> 4;
                int c = (v & 15) * 8;
                breg[i] = *(const uint4*)(Bm + (long long)(kn + r) * ldb + (n0 + c));
            }
            {
                int v = tid;              // one cacheline probe per thread
                int r = v >> 3;
                int c = (v & 7) * 8;
                __builtin_prefetch(A + (long long)(rowBase + r) * lda + (kn + c), 0, 1);
            }
        }

        // own async A transfers must be complete before the group barrier
        asm volatile("s_wait_asynccnt 0x0" ::: "memory");
        __syncthreads();

        #pragma unroll
        for (int kk = 0; kk < TILE_K; kk += 32) {
            FragBF af[2], bfr[4];
            // A frag: lane holds K runs [g*8, g*8+8) and [16+g*8, 16+g*8+8)
            #pragma unroll
            for (int i = 0; i < 2; ++i) {
                const __bf16* p = &As[(wm * 32 + i * 16 + lr) * LDSS + kk];
                af[i].h[0] = *(const v8bf*)(p + g * 8);
                af[i].h[1] = *(const v8bf*)(p + 16 + g * 8);
            }
            // B frag: lane holds K runs [g*8, g*8+8) and [16+g*8, 16+g*8+8) of column n (transposed LDS)
            #pragma unroll
            for (int j = 0; j < 4; ++j) {
                const __bf16* p = &Bs[(wn * 64 + j * 16 + lr) * LDSS + kk];
                bfr[j].h[0] = *(const v8bf*)(p + g * 8);
                bfr[j].h[1] = *(const v8bf*)(p + 16 + g * 8);
            }
            #pragma unroll
            for (int i = 0; i < 2; ++i)
                #pragma unroll
                for (int j = 0; j < 4; ++j)
                    acc[i][j] = __builtin_amdgcn_wmma_f32_16x16x32_bf16(
                        false, af[i].v, false, bfr[j].v, (short)0, acc[i][j],
                        false, false);
        }
    }

    // --- epilogue: C/D layout: M = vgpr + 8*g, N = lane%16 ---
    #pragma unroll
    for (int i = 0; i < 2; ++i) {
        int rbase = rowBase + wm * 32 + i * 16 + g * 8;
        float scv[8];
        #pragma unroll
        for (int v = 0; v < 8; ++v)
            scv[v] = rs ? rs[rbase + v] * scalemul : scalemul;
        #pragma unroll
        for (int j = 0; j < 4; ++j) {
            int col = n0 + wn * 64 + j * 16 + lr;
            #pragma unroll
            for (int v = 0; v < 8; ++v) {
                int row  = rbase + v;
                float c  = acc[i][j][v];
                long long idx = cOff + (long long)row * ldc + col;
                if (epi == 0) {
                    outF[idx] = c * scv[v];
                } else if (epi == 1) {
                    outB[idx] = (__bf16)fmaxf(c, 0.0f);
                } else if (epi == 2) {
                    // elu(c)+1 == exp(c) for c<=0, c+1 for c>0
                    outB[idx]  = (__bf16)(c > 0.0f ? c + 1.0f : __expf(c));
                    outB2[idx] = (__bf16)(c * invL);
                } else {
                    outB[idx] = (__bf16)(c * scv[v]);
                }
            }
        }
    }
}

// ---------------------------------------------------------------------------
// KV kernel: KV[b,h,i,j] = sum_l feat[b,l,h*128+i] * V[b,l,h*128+j]
// = feat^T @ V per (b,h). Contraction over L; both operands staged transposed.
// Partial sums over an L-chunk, f32 atomics into KV (KV pre-zeroed).
// grid = (L/Lchunk, B*H), block = 256.
// ---------------------------------------------------------------------------
__global__ __launch_bounds__(256) void kv_kernel(
    const __bf16* __restrict__ feat, const __bf16* __restrict__ vmat,
    float* __restrict__ KV, int L, int D, int H, int Lchunk)
{
    __shared__ __bf16 As[128 * LDSS];  // feat^T tile [i][l]
    __shared__ __bf16 Bs[128 * LDSS];  // V^T    tile [j][l]

    const int tid  = threadIdx.x;
    const int lane = tid & 31;
    const int wid  = tid >> 5;
    const int wm   = wid & 3;
    const int wn   = wid >> 2;
    const int lr   = lane & 15;
    const int g    = lane >> 4;

    const int bh = blockIdx.y;
    const int b  = bh / H;
    const int h  = bh - b * H;
    const long long base = (long long)b * L * D + (long long)h * 128;
    const int l0 = blockIdx.x * Lchunk;

    v8f acc[2][4] = {};

    for (int lc = 0; lc < Lchunk; lc += TILE_K) {
        #pragma unroll
        for (int i = 0; i < 4; ++i) {
            int v = tid + i * 256;
            int r = v >> 4;                   // l index 0..63
            int c = (v & 15) * 8;             // d index 0..120
            long long rowoff = base + (long long)(l0 + lc + r) * D + c;
            uint4 dA = *(const uint4*)(feat + rowoff);
            uint4 dB = *(const uint4*)(vmat + rowoff);
            const __bf16* ea = (const __bf16*)&dA;
            const __bf16* eb = (const __bf16*)&dB;
            #pragma unroll
            for (int j = 0; j < 8; ++j) {
                As[(c + j) * LDSS + r] = ea[j];
                Bs[(c + j) * LDSS + r] = eb[j];
            }
        }
        __syncthreads();

        #pragma unroll
        for (int kk = 0; kk < TILE_K; kk += 32) {
            FragBF af[2], bfr[4];
            #pragma unroll
            for (int i = 0; i < 2; ++i) {
                const __bf16* p = &As[(wm * 32 + i * 16 + lr) * LDSS + kk];
                af[i].h[0] = *(const v8bf*)(p + g * 8);
                af[i].h[1] = *(const v8bf*)(p + 16 + g * 8);
            }
            #pragma unroll
            for (int j = 0; j < 4; ++j) {
                const __bf16* p = &Bs[(wn * 64 + j * 16 + lr) * LDSS + kk];
                bfr[j].h[0] = *(const v8bf*)(p + g * 8);
                bfr[j].h[1] = *(const v8bf*)(p + 16 + g * 8);
            }
            #pragma unroll
            for (int i = 0; i < 2; ++i)
                #pragma unroll
                for (int j = 0; j < 4; ++j)
                    acc[i][j] = __builtin_amdgcn_wmma_f32_16x16x32_bf16(
                        false, af[i].v, false, bfr[j].v, (short)0, acc[i][j],
                        false, false);
        }
        __syncthreads();
    }

    float* kvb = KV + (long long)bh * 128 * 128;
    #pragma unroll
    for (int i = 0; i < 2; ++i) {
        int rbase = wm * 32 + i * 16 + g * 8;
        #pragma unroll
        for (int j = 0; j < 4; ++j) {
            int col = wn * 64 + j * 16 + lr;
            #pragma unroll
            for (int v = 0; v < 8; ++v)
                atomicAdd(&kvb[(rbase + v) * 128 + col], acc[i][j][v]);
        }
    }
}

// Ksum[b*D + c] = sum_l feat[b,l,c]   (atomic partials over L chunks)
__global__ void ksum_kernel(const __bf16* __restrict__ feat, float* __restrict__ Ksum,
                            int L, int D, int nchunk)
{
    int c = blockIdx.x * blockDim.x + threadIdx.x;   // [0, B*D)
    int b = c / D;
    int lchunk = L / nchunk;
    long long p = (long long)b * L * D + (long long)(blockIdx.y * lchunk) * D + (c - b * D);
    float s = 0.0f;
    for (int l = 0; l < lchunk; ++l) { s += (float)feat[p]; p += D; }
    atomicAdd(&Ksum[c], s);
}

// Zt[(b*H+h)*L + l] = 1 / (dot(feat[b,l,h*128:], Ksum[b,h*128:]) + eps)
// one wave per (b,l,h)
__global__ void z_kernel(const __bf16* __restrict__ feat, const float* __restrict__ Ksum,
                         float* __restrict__ Zt, int L, int D, int H, long long total)
{
    long long w = ((long long)blockIdx.x * blockDim.x + threadIdx.x) >> 5;
    int lane = threadIdx.x & 31;
    if (w >= total) return;
    int h = (int)(w % H);
    long long t = w / H;
    int l = (int)(t % L);
    int b = (int)(t / L);
    int d = D / H;
    const __bf16* q = feat + ((long long)b * L + l) * D + (long long)h * d;
    const float* ks = Ksum + (long long)b * D + (long long)h * d;
    float s = 0.0f;
    for (int j = lane; j < d; j += 32) s += (float)q[j] * ks[j];
    #pragma unroll
    for (int off = 16; off > 0; off >>= 1) s += __shfl_down(s, off, 32);
    if (lane == 0) Zt[((long long)b * H + h) * L + l] = 1.0f / (s + 1e-6f);
}

// out[row,:] = base[row,:] + LayerNorm(y[row,:]) * g + b   (D = 1024, block 256)
// base may alias out (each element is read by its owning thread before write).
__global__ __launch_bounds__(256) void ln_add_kernel(
    const float* base, const float* __restrict__ y,
    const float* __restrict__ g, const float* __restrict__ bb,
    float* out, int D)
{
    long long row = blockIdx.x;
    const float* yr = y + row * D;
    int tid = threadIdx.x;
    float v[4];
    float s = 0.0f, s2 = 0.0f;
    #pragma unroll
    for (int k = 0; k < 4; ++k) {
        float t = yr[tid + k * 256];
        v[k] = t; s += t; s2 += t * t;
    }
    #pragma unroll
    for (int off = 16; off > 0; off >>= 1) {
        s  += __shfl_down(s,  off, 32);
        s2 += __shfl_down(s2, off, 32);
    }
    __shared__ float ls[8], ls2[8];
    __shared__ float mu_s, rstd_s;
    int lane = tid & 31, wid = tid >> 5;
    if (lane == 0) { ls[wid] = s; ls2[wid] = s2; }
    __syncthreads();
    if (tid == 0) {
        float S = 0.0f, S2 = 0.0f;
        for (int i = 0; i < 8; ++i) { S += ls[i]; S2 += ls2[i]; }
        float mu  = S / D;
        float var = S2 / D - mu * mu;   // biased variance (matches jnp.var)
        mu_s = mu; rstd_s = rsqrtf(var + 1e-5f);
    }
    __syncthreads();
    float mu = mu_s, r = rstd_s;
    #pragma unroll
    for (int k = 0; k < 4; ++k) {
        int c = tid + k * 256;
        out[row * D + c] = base[row * D + c] + (v[k] - mu) * r * g[c] + bb[c];
    }
}

// f32 -> bf16 elementwise convert (grid-stride)
__global__ void f2bf_kernel(const float* __restrict__ in, __bf16* __restrict__ o, long long n)
{
    long long i = (long long)blockIdx.x * blockDim.x + threadIdx.x;
    long long stride = (long long)gridDim.x * blockDim.x;
    for (; i < n; i += stride) o[i] = (__bf16)in[i];
}

// ---------------------------------------------------------------------------
extern "C" void kernel_launch(void* const* d_in, const int* in_sizes, int n_in,
                              void* d_out, int out_size, void* d_ws, size_t ws_size,
                              hipStream_t stream)
{
    (void)in_sizes; (void)n_in; (void)out_size; (void)ws_size;

    const float* x  = (const float*)d_in[0];
    const float* Wq = (const float*)d_in[1];
    const float* Wm = (const float*)d_in[2];
    const float* W1 = (const float*)d_in[3];
    const float* W2 = (const float*)d_in[4];
    const float* g1 = (const float*)d_in[5];
    const float* b1 = (const float*)d_in[6];
    const float* g2 = (const float*)d_in[7];
    const float* b2 = (const float*)d_in[8];
    float* out = (float*)d_out;
    char* ws = (char*)d_ws;

    const long long L = C_L, D = C_D, Bn = C_B, H = C_H;
    const long long LD  = L * D;            // 8388608
    const long long BLD = Bn * LD;          // 16777216
    const long long DD  = D * D;            // 1048576
    const long long KVN = Bn * H * 128 * 128; // 262144

    // ---- workspace layout (aliased; ~216 MB) ----
    size_t off = 0;
    char* p_hb   = ws + off; off += (size_t)(4 * BLD);   // h (B,L,2D) bf16 = 67MB; first half aliased as x_bf16
    char* p_feat = ws + off; off += (size_t)(2 * BLD);   // feat bf16
    char* p_msg  = ws + off; off += (size_t)(2 * BLD);   // msg bf16
    char* p_scr  = ws + off; off += (size_t)(4 * BLD);   // f32 scratch (y1/mlp); first half aliased as V bf16
    char* p_wq   = ws + off; off += (size_t)(2 * DD);
    char* p_wm   = ws + off; off += (size_t)(2 * DD);
    char* p_w1   = ws + off; off += (size_t)(4 * DD);
    char* p_w2   = ws + off; off += (size_t)(4 * DD);
    char* p_kvf  = ws + off; off += (size_t)(4 * KVN);
    char* p_kvb  = ws + off; off += (size_t)(2 * KVN);
    char* p_ks   = ws + off; off += (size_t)(4 * Bn * D);
    char* p_z    = ws + off; off += (size_t)(4 * Bn * H * L);

    __bf16* hb    = (__bf16*)p_hb;
    __bf16* xb    = (__bf16*)p_hb;     // alias: dead before hb is written
    __bf16* featb = (__bf16*)p_feat;
    __bf16* msgb  = (__bf16*)p_msg;
    float*  scrF  = (float*)p_scr;
    __bf16* vb    = (__bf16*)p_scr;    // alias: dead before scrF is written
    __bf16* wqb   = (__bf16*)p_wq;
    __bf16* wmb   = (__bf16*)p_wm;
    __bf16* w1b   = (__bf16*)p_w1;
    __bf16* w2b   = (__bf16*)p_w2;
    float*  KVf   = (float*)p_kvf;
    __bf16* KVb   = (__bf16*)p_kvb;
    float*  Ksum  = (float*)p_ks;
    float*  Zt    = (float*)p_z;

    dim3 blk(256);

    // 1) convert inputs to bf16
    f2bf_kernel<<<2048, blk, 0, stream>>>(x,  xb,  BLD);
    f2bf_kernel<<<512,  blk, 0, stream>>>(Wq, wqb, DD);
    f2bf_kernel<<<512,  blk, 0, stream>>>(Wm, wmb, DD);
    f2bf_kernel<<<1024, blk, 0, stream>>>(W1, w1b, 2 * DD);
    f2bf_kernel<<<1024, blk, 0, stream>>>(W2, w2b, 2 * DD);

    // 2) zero atomic accumulators
    hipMemsetAsync(KVf,  0, (size_t)(4 * KVN), stream);
    hipMemsetAsync(Ksum, 0, (size_t)(4 * Bn * D), stream);

    // 3) q = x@Wq fused: feat = elu(q)+1 (bf16), V = q/L (bf16)
    gemm_bf16_kernel<<<dim3(128, 8, 1), blk, 0, stream>>>(
        xb, 1024, 0, 0, wqb, 1024, 0,
        nullptr, featb, vb, 1024, 0, 0,
        nullptr, 0, 1.0f, 1.0f / 8192.0f, 1, 1024, /*epi=*/2);

    // 4) KV = feat^T @ V per (b,h)
    kv_kernel<<<dim3(16, 16), blk, 0, stream>>>(featb, vb, KVf, (int)L, (int)D, (int)H, 512);

    // 5) Ksum = sum_l feat
    ksum_kernel<<<dim3(8, 32), blk, 0, stream>>>(featb, Ksum, (int)L, (int)D, 32);

    // 6) KV -> bf16
    f2bf_kernel<<<256, blk, 0, stream>>>(KVf, KVb, KVN);

    // 7) Z = 1/(Q.Ksum + eps), stored [B,H,L]
    z_kernel<<<16384, blk, 0, stream>>>(featb, Ksum, Zt, (int)L, (int)D, (int)H, Bn * L * H);

    // 8) msg = (Q @ KV) * Z * L, batched over z = b*H + h
    gemm_bf16_kernel<<<dim3(64, 1, 16), blk, 0, stream>>>(
        featb, 1024, LD, 128, KVb, 128, 128 * 128,
        nullptr, msgb, nullptr, 1024, LD, 128,
        Zt, L, 8192.0f, 0.0f, /*HH=*/8, 128, /*epi=*/3);

    // 9) y1 = msg @ Wm  (f32)   [overwrites vb alias - dead]
    gemm_bf16_kernel<<<dim3(128, 8, 1), blk, 0, stream>>>(
        msgb, 1024, 0, 0, wmb, 1024, 0,
        scrF, nullptr, nullptr, 1024, 0, 0,
        nullptr, 0, 1.0f, 0.0f, 1, 1024, /*epi=*/0);

    // 10) out = x + LN(y1)
    ln_add_kernel<<<16384, blk, 0, stream>>>(x, scrF, g1, b1, out, (int)D);

    // 11) h = relu(msg @ W1) (bf16)   [overwrites xb alias - dead]
    gemm_bf16_kernel<<<dim3(128, 16, 1), blk, 0, stream>>>(
        msgb, 1024, 0, 0, w1b, 2048, 0,
        nullptr, hb, nullptr, 2048, 0, 0,
        nullptr, 0, 1.0f, 0.0f, 1, 1024, /*epi=*/1);

    // 12) mlp = h @ W2 (f32)
    gemm_bf16_kernel<<<dim3(128, 8, 1), blk, 0, stream>>>(
        hb, 2048, 0, 0, w2b, 1024, 0,
        scrF, nullptr, nullptr, 1024, 0, 0,
        nullptr, 0, 1.0f, 0.0f, 1, 2048, /*epi=*/0);

    // 13) out = out + LN(mlp)
    ln_add_kernel<<<16384, blk, 0, stream>>>(out, scrF, g2, b2, out, (int)D);
}
